// QuaxConvGeneralDilated_82343112999366
// MI455X (gfx1250) — 8-bit matrix operand order repaired and checked by reading, compiled, not run on hardware
//
#include <hip/hip_runtime.h>
#include <stdint.h>

typedef __attribute__((ext_vector_type(8))) int   v8i;
typedef __attribute__((ext_vector_type(4))) float v4f;

#define B_   32
#define H_   112
#define W_   112
#define CI_  128
#define CO_  256
#define KTOT 1152

__device__ __forceinline__ float wave_red_max(float v) {
    #pragma unroll
    for (int off = 16; off > 0; off >>= 1)
        v = fmaxf(v, __shfl_xor(v, off, 32));
    return v;
}

__global__ void k_init_amax(float* amax) {
    if (threadIdx.x < B_) amax[threadIdx.x] = 0.0f;
}

__global__ void k_lhs_absmax(const float* __restrict__ lhs, float* __restrict__ amax) {
    const int slices = 64;
    int b = blockIdx.x / slices;
    int s = blockIdx.x % slices;
    const size_t per_batch = (size_t)H_ * W_ * CI_;
    const size_t per_slice = per_batch / slices;
    const v4f* p = (const v4f*)(lhs + (size_t)b * per_batch + (size_t)s * per_slice);
    size_t n4 = per_slice / 4;
    float m = 0.0f;
    for (size_t i = threadIdx.x; i < n4; i += blockDim.x) {
        v4f v = __builtin_nontemporal_load(&p[i]);
        m = fmaxf(m, fmaxf(fmaxf(fabsf(v.x), fabsf(v.y)), fmaxf(fabsf(v.z), fabsf(v.w))));
    }
    m = wave_red_max(m);
    __shared__ float sm[8];
    int lane = threadIdx.x & 31, w = threadIdx.x >> 5;
    if (lane == 0) sm[w] = m;
    __syncthreads();
    if (w == 0) {
        m = (lane < (int)(blockDim.x >> 5)) ? sm[lane] : 0.0f;
        m = wave_red_max(m);
        if (lane == 0) atomicMax((unsigned int*)&amax[b], __float_as_uint(m));
    }
}

__global__ void k_lhs_scales(const float* __restrict__ amax,
                             float* __restrict__ s_lhs, float* __restrict__ inv_lhs) {
    int i = threadIdx.x;
    if (i < B_) {
        float a = amax[i];
        float s = (a > 0.0f) ? (a * (1.0f / 127.0f)) : 1.0f;
        s_lhs[i] = s;
        inv_lhs[i] = 1.0f / s;
    }
}

__device__ __forceinline__ signed char q8(float x, float inv) {
    float r = rintf(x * inv);
    r = fminf(127.0f, fmaxf(-127.0f, r));
    return (signed char)(int)r;
}

__global__ void k_lhs_quant(const float* __restrict__ lhs, const float* __restrict__ inv_lhs,
                            signed char* __restrict__ q) {
    size_t i4 = (size_t)blockIdx.x * blockDim.x + threadIdx.x;
    const size_t total4 = (size_t)B_ * H_ * W_ * CI_ / 4;
    if (i4 >= total4) return;
    const size_t per_b4 = (size_t)H_ * W_ * CI_ / 4;
    float inv = inv_lhs[(int)(i4 / per_b4)];
    v4f v = __builtin_nontemporal_load(&((const v4f*)lhs)[i4]);
    char4 c;
    c.x = q8(v.x, inv); c.y = q8(v.y, inv); c.z = q8(v.z, inv); c.w = q8(v.w, inv);
    ((char4*)q)[i4] = c;
}

__global__ void k_rhs_quant(const float* __restrict__ rhs, float* __restrict__ s_rhs,
                            signed char* __restrict__ qT) {
    int co = blockIdx.x;
    float m = 0.0f;
    for (int k = threadIdx.x; k < KTOT; k += blockDim.x)
        m = fmaxf(m, fabsf(rhs[(size_t)k * CO_ + co]));
    m = wave_red_max(m);
    __shared__ float sm[8];
    __shared__ float s_inv;
    int lane = threadIdx.x & 31, w = threadIdx.x >> 5;
    if (lane == 0) sm[w] = m;
    __syncthreads();
    if (threadIdx.x == 0) {
        float mm = 0.0f;
        for (int i = 0; i < (int)(blockDim.x >> 5); ++i) mm = fmaxf(mm, sm[i]);
        float s = (mm > 0.0f) ? (mm * (1.0f / 127.0f)) : 1.0f;
        s_rhs[co] = s;
        s_inv = 1.0f / s;
    }
    __syncthreads();
    float inv = s_inv;
    for (int k = threadIdx.x; k < KTOT; k += blockDim.x)
        qT[(size_t)co * KTOT + k] = q8(rhs[(size_t)k * CO_ + co], inv);
}

__global__ __launch_bounds__(256) void k_conv_wmma_iu8(
    const signed char* __restrict__ qlhs, const signed char* __restrict__ qrhsT,
    const float* __restrict__ s_lhs, const float* __restrict__ s_rhs,
    float* __restrict__ out)
{
    const int XT = W_ / 16;
    int tile = blockIdx.x;
    int b  = tile / (H_ * XT);
    int r  = tile % (H_ * XT);
    int y  = r / XT;
    int x0 = (r % XT) * 16;

    int tid  = threadIdx.x;
    int wave = tid >> 5;
    int lane = tid & 31;
    int lm   = lane & 15;
    int hi   = lane >> 4;

    int co0 = (wave * 2 + 0) * 16;
    int co1 = (wave * 2 + 1) * 16;

    const signed char* brow0 = qrhsT + (size_t)(co0 + lm) * KTOT + hi * 16;
    const signed char* brow1 = qrhsT + (size_t)(co1 + lm) * KTOT + hi * 16;

    const int akoff = hi * 16;

    v8i acc0 = {};
    v8i acc1 = {};

    for (int tap = 0; tap < 9; ++tap) {
        int ky = tap / 3, kx = tap % 3;
        int yin = y + ky - 1;
        int xin = x0 + lm + kx - 1;
        bool valid = (yin >= 0) && (yin < H_) && (xin >= 0) && (xin < W_);
        int yc = min(max(yin, 0), H_ - 1);
        int xc = min(max(xin, 0), W_ - 1);
        const signed char* abase =
            qlhs + (((size_t)b * H_ + yc) * W_ + xc) * CI_ + akoff;
        #pragma unroll
        for (int half = 0; half < 2; ++half) {
            int ci0 = half * 64;
            const int* ap = (const int*)(abase + ci0);
            int2 p0 = *(const int2*)(ap + 0);
            int2 p1 = *(const int2*)(ap + 2);
            int2 p2 = *(const int2*)(ap + 8);
            int2 p3 = *(const int2*)(ap + 10);
            if (!valid) {
                p0 = make_int2(0, 0); p1 = make_int2(0, 0);
                p2 = make_int2(0, 0); p3 = make_int2(0, 0);
            }
            v8i a = { p0.x, p0.y, p1.x, p1.y, p2.x, p2.y, p3.x, p3.y };

            int kbase = tap * 128 + ci0;
            const int4* bp0 = (const int4*)(brow0 + kbase);
            const int4* bp1 = (const int4*)(brow1 + kbase);
            int4 q00 = bp0[0], q01 = bp0[2];
            int4 q10 = bp1[0], q11 = bp1[2];
            v8i bb0 = { q00.x, q00.y, q00.z, q00.w, q01.x, q01.y, q01.z, q01.w };
            v8i bb1 = { q10.x, q10.y, q10.z, q10.w, q11.x, q11.y, q11.z, q11.w };

            acc0 = __builtin_amdgcn_wmma_i32_16x16x64_iu8(true, a, true, bb0, acc0, false, false);
            acc1 = __builtin_amdgcn_wmma_i32_16x16x64_iu8(true, a, true, bb1, acc1, false, false);
        }
    }

    float sl  = s_lhs[b];
    float sr0 = s_rhs[co0 + lm] * sl;
    float sr1 = s_rhs[co1 + lm] * sl;
    size_t outbase = (((size_t)b * H_ + y) * W_ + x0) * CO_;
    #pragma unroll
    for (int v = 0; v < 8; ++v) {
        int row = v + hi * 8;
        size_t o = outbase + (size_t)row * CO_;
        __builtin_nontemporal_store((float)acc0[v] * sr0, &out[o + co0 + lm]);
        __builtin_nontemporal_store((float)acc1[v] * sr1, &out[o + co1 + lm]);
    }
}

extern "C" void kernel_launch(void* const* d_in, const int* in_sizes, int n_in,
                              void* d_out, int out_size, void* d_ws, size_t ws_size,
                              hipStream_t stream) {
    (void)in_sizes; (void)n_in; (void)out_size; (void)ws_size;
    const float* lhs = (const float*)d_in[0];
    const float* rhs = (const float*)d_in[1];
    float* out = (float*)d_out;

    uint8_t* ws = (uint8_t*)d_ws;
    float*       amax    = (float*)(ws + 0);
    float*       s_lhs   = (float*)(ws + 256);
    float*       inv_lhs = (float*)(ws + 512);
    float*       s_rhs   = (float*)(ws + 768);
    signed char* qrhsT   = (signed char*)(ws + 2048);
    signed char* qlhs    = (signed char*)(ws + 2048 + (size_t)CO_ * KTOT);

    k_init_amax<<<1, 32, 0, stream>>>(amax);
    k_lhs_absmax<<<B_ * 64, 256, 0, stream>>>(lhs, amax);
    k_lhs_scales<<<1, 32, 0, stream>>>(amax, s_lhs, inv_lhs);

    size_t total4 = (size_t)B_ * H_ * W_ * CI_ / 4;
    k_lhs_quant<<<(unsigned)((total4 + 255) / 256), 256, 0, stream>>>(lhs, inv_lhs, qlhs);
    k_rhs_quant<<<CO_, 256, 0, stream>>>(rhs, s_rhs, qrhsT);

    k_conv_wmma_iu8<<<B_ * H_ * (W_ / 16), 256, 0, stream>>>(qlhs, qrhsT, s_lhs, s_rhs, out);
}
